// Attention_29265907155069
// MI455X (gfx1250) — matrix operand order repaired and checked by reading, not compiled, not run on hardware
//
#include <hip/hip_runtime.h>
#include <hip/hip_bf16.h>

typedef __attribute__((ext_vector_type(16))) _Float16 v16h;
typedef __attribute__((ext_vector_type(8)))  float    v8f;

#define B_ 4
#define N_ 2048
#define C_ 768
#define H_ 12
#define D_ 64
#define SCALE_ 64.0f
#define MASK_NEG_ -1000000.0f

static __device__ __forceinline__ v8f wmma_f16(v16h a, v16h b, v8f c) {
  // 8 args: (neg_a, A, neg_b, B, c_mod, C, reuse_a, reuse_b)
  return __builtin_amdgcn_wmma_f32_16x16x32_f16(false, a, false, b, (short)0, c,
                                                false, false);
}

static __device__ __forceinline__ void wait_async0() {
#if __has_builtin(__builtin_amdgcn_s_wait_asynccnt)
  __builtin_amdgcn_s_wait_asynccnt(0);
#else
  asm volatile("s_wait_asynccnt 0x0" ::: "memory");
#endif
}

// Copy one 128-byte row per lane, global -> LDS, via CDNA5 async-LDS path.
// INST_OFFSET is added to BOTH the global and LDS addresses (ISA 15.18.3),
// so one LDS-dest VGPR + one 64-bit global address covers the whole row.
static __device__ __forceinline__ void async_copy_row128(unsigned lds_off,
                                                         const void* gsrc) {
  unsigned long long ga = (unsigned long long)gsrc;
  asm volatile(
      "global_load_async_to_lds_b128 %0, %1, off\n\t"
      "global_load_async_to_lds_b128 %0, %1, off offset:16\n\t"
      "global_load_async_to_lds_b128 %0, %1, off offset:32\n\t"
      "global_load_async_to_lds_b128 %0, %1, off offset:48\n\t"
      "global_load_async_to_lds_b128 %0, %1, off offset:64\n\t"
      "global_load_async_to_lds_b128 %0, %1, off offset:80\n\t"
      "global_load_async_to_lds_b128 %0, %1, off offset:96\n\t"
      "global_load_async_to_lds_b128 %0, %1, off offset:112"
      :: "v"(lds_off), "v"(ga) : "memory");
}

// ---------------------------------------------------------------------------
// Kernel 1: qkv = x @ qkv_w^T  ->  q/k/v f16 buffers in [B,H,N,D] layout
// One wave computes a 16x64 output tile. K-loop step 32 (WMMA K).
// ---------------------------------------------------------------------------
__global__ __launch_bounds__(128) void qkv_gemm(const float* __restrict__ x,
                                                const float* __restrict__ w,
                                                _Float16* __restrict__ qkv_out) {
  const int lane = threadIdx.x & 31;
  const int wid  = threadIdx.x >> 5;
  const int gw   = blockIdx.x * 4 + wid;
  const int MT = (B_ * N_) / 16;   // 512
  const int CT = (3 * C_) / 64;    // 36
  if (gw >= MT * CT) return;
  const int mt = gw % MT, ct = gw / MT;
  const int m0 = mt * 16, n0 = ct * 64;
  const int half = lane >> 4, l16 = lane & 15;

  v8f acc[4] = {};
  for (int k0 = 0; k0 < C_; k0 += 32) {
    // A fragment: row = m0+l16 ; K = k0 + half*16 .. +15 (contiguous)
    const float* ar = x + (size_t)(m0 + l16) * C_ + k0 + half * 16;
    v16h a;
#pragma unroll
    for (int i = 0; i < 8; ++i) {
      a[i]     = (_Float16)ar[i];
      a[i + 8] = (_Float16)ar[i + 8];
    }
#pragma unroll
    for (int t = 0; t < 4; ++t) {
      // B fragment: col = n0+t*16+l16 ; K = k0 + half*16 .. +15 (contig in W row)
      const float* br = w + (size_t)(n0 + t * 16 + l16) * C_ + k0 + half * 16;
      v16h b;
#pragma unroll
      for (int i = 0; i < 16; ++i) b[i] = (_Float16)br[i];
      acc[t] = wmma_f16(a, b, acc[t]);
    }
  }

  // scatter to q/k/v f16: which = n0/768, head = (n0%768)/64, d = t*16+l16
  const int which = n0 / C_;
  const int h     = (n0 % C_) / D_;
  const int b     = m0 >> 11;        // / N_
  const int nrow  = m0 & (N_ - 1);
  _Float16* dst = qkv_out + (size_t)which * ((size_t)B_ * H_ * N_ * D_) +
                  (size_t)(b * H_ + h) * N_ * D_;
#pragma unroll
  for (int t = 0; t < 4; ++t)
#pragma unroll
    for (int r = 0; r < 8; ++r)
      dst[(size_t)(nrow + r + 8 * half) * D_ + t * 16 + l16] =
          (_Float16)acc[t][r];
}

// ---------------------------------------------------------------------------
// Kernel 2: flash-attention. One wave per (b,h, 16-query tile).
// Loop over 64 chunks of 32 keys: S = Q K^T (4 WMMAs), scale*64 + mask*-1e6,
// online softmax (f32 stats), P -> LDS -> A-fragment, PV (4 WMMAs).
// V chunk staged to LDS via global_load_async_to_lds_b128 (ASYNCcnt).
// ---------------------------------------------------------------------------
__global__ __launch_bounds__(128) void attn_kernel(
    const _Float16* __restrict__ qh, const _Float16* __restrict__ kh,
    const _Float16* __restrict__ vh, const int* __restrict__ mask,
    _Float16* __restrict__ ao) {
  __shared__ __align__(16) _Float16 lds_p[4][16 * 32];
  __shared__ __align__(16) _Float16 lds_v[4][32 * 64];  // row pitch 64 halfs

  const int lane = threadIdx.x & 31;
  const int wid  = threadIdx.x >> 5;
  const int gw   = blockIdx.x * 4 + wid;
  const int QT   = N_ / 16;  // 128
  if (gw >= B_ * H_ * QT) return;
  const int qt = gw % QT, bh = gw / QT;
  const int b = bh / H_, h = bh % H_;
  const int half = lane >> 4, l16 = lane & 15;
  const int q0 = qt * 16;

  const _Float16* qp = qh + (size_t)bh * N_ * D_;
  const _Float16* kp = kh + (size_t)bh * N_ * D_;
  const _Float16* vp = vh + (size_t)bh * N_ * D_;

  // per-lane LDS byte offset of this wave's V row (generic ptr low 32 bits =
  // LDS address per the aperture mapping)
  const unsigned lds_v_row =
      (unsigned)(size_t)(&lds_v[wid][0]) + (unsigned)lane * (D_ * 2);

  // Q tile as two A fragments: K-dim tiles d=[0,32) and d=[32,64)
  const _Float16* qr = qp + (size_t)(q0 + l16) * D_ + half * 16;
  v16h aq0, aq1;
#pragma unroll
  for (int i = 0; i < 8; ++i) {
    aq0[i] = qr[i];      aq0[i + 8] = qr[i + 8];
    aq1[i] = qr[i + 32]; aq1[i + 8] = qr[i + 40];
  }

  float mrow[8], lrow[8];
  v8f o[4] = {};
#pragma unroll
  for (int r = 0; r < 8; ++r) { mrow[r] = -3.0e38f; lrow[r] = 0.0f; }

  for (int kb = 0; kb < N_; kb += 32) {
    // async stage of V chunk (32 rows x 128B) into LDS; each lane one row.
    // WAR vs previous iteration's ds reads is safe: the PV WMMA consuming
    // bv forces s_wait_dscnt before we get here again.
    async_copy_row128(lds_v_row, vp + (size_t)(kb + lane) * D_);

    // S = Q K^T : two 16x16 score tiles (key groups of 16), K(=D)=64 -> 2 WMMA each
    v8f s0 = {}, s1 = {};
    {
      const _Float16* kr0 = kp + (size_t)(kb + l16) * D_ + half * 16;
      const _Float16* kr1 = kp + (size_t)(kb + 16 + l16) * D_ + half * 16;
      v16h bk;
#pragma unroll
      for (int i = 0; i < 16; ++i) bk[i] = kr0[i];
      s0 = wmma_f16(aq0, bk, s0);
#pragma unroll
      for (int i = 0; i < 16; ++i) bk[i] = kr0[i + 32];
      s0 = wmma_f16(aq1, bk, s0);
#pragma unroll
      for (int i = 0; i < 16; ++i) bk[i] = kr1[i];
      s1 = wmma_f16(aq0, bk, s1);
#pragma unroll
      for (int i = 0; i < 16; ++i) bk[i] = kr1[i + 32];
      s1 = wmma_f16(aq1, bk, s1);
    }

    // scale + mask + online softmax update (C/D layout: row = r + 8*half, col = l16)
    const int kcol0 = kb + l16, kcol1 = kb + 16 + l16;
    float p0[8], p1[8], alpha[8];
#pragma unroll
    for (int r = 0; r < 8; ++r) {
      const int qrow = q0 + r + 8 * half;
      float x0 = s0[r] * SCALE_ + (float)mask[(size_t)qrow * N_ + kcol0] * MASK_NEG_;
      float x1 = s1[r] * SCALE_ + (float)mask[(size_t)qrow * N_ + kcol1] * MASK_NEG_;
      float tm = fmaxf(x0, x1);
      tm = fmaxf(tm, __shfl_xor(tm, 1));
      tm = fmaxf(tm, __shfl_xor(tm, 2));
      tm = fmaxf(tm, __shfl_xor(tm, 4));
      tm = fmaxf(tm, __shfl_xor(tm, 8));
      float mnew = fmaxf(mrow[r], tm);
      alpha[r] = __expf(mrow[r] - mnew);
      mrow[r] = mnew;
      p0[r] = __expf(x0 - mnew);
      p1[r] = __expf(x1 - mnew);
      float ts = p0[r] + p1[r];
      ts += __shfl_xor(ts, 1);
      ts += __shfl_xor(ts, 2);
      ts += __shfl_xor(ts, 4);
      ts += __shfl_xor(ts, 8);
      lrow[r] = lrow[r] * alpha[r] + ts;
    }
#pragma unroll
    for (int t = 0; t < 4; ++t)
#pragma unroll
      for (int r = 0; r < 8; ++r) o[t][r] *= alpha[r];

    // P (16x32 f16) via LDS: C/D layout -> A layout transpose across lanes
#pragma unroll
    for (int r = 0; r < 8; ++r) {
      lds_p[wid][(r + 8 * half) * 32 + l16]      = (_Float16)p0[r];
      lds_p[wid][(r + 8 * half) * 32 + 16 + l16] = (_Float16)p1[r];
    }
    v16h pa;
#pragma unroll
    for (int i = 0; i < 8; ++i) {
      pa[i]     = lds_p[wid][l16 * 32 + half * 16 + i];
      pa[i + 8] = lds_p[wid][l16 * 32 + half * 16 + 8 + i];
    }

    // wait for async V staging, then O += P V (4 d-tiles of 16 cols)
    wait_async0();
#pragma unroll
    for (int t = 0; t < 4; ++t) {
      v16h bv;
#pragma unroll
      for (int i = 0; i < 16; ++i)
        bv[i] = lds_v[wid][(half * 16 + i) * D_ + t * 16 + l16];
      o[t] = wmma_f16(pa, bv, o[t]);
    }
  }

  // epilogue: normalize, write f16 attn-out in [B,N,C] layout
  _Float16* aop = ao + (size_t)b * N_ * C_ + h * D_;
#pragma unroll
  for (int r = 0; r < 8; ++r) {
    const float inv = 1.0f / lrow[r];
    const int row = q0 + r + 8 * half;
#pragma unroll
    for (int t = 0; t < 4; ++t)
      aop[(size_t)row * C_ + t * 16 + l16] = (_Float16)(o[t][r] * inv);
  }
}

// ---------------------------------------------------------------------------
// Kernel 3: out = attn_out @ proj_w^T + proj_b  (f32 output)
// ---------------------------------------------------------------------------
__global__ __launch_bounds__(128) void proj_gemm(const _Float16* __restrict__ ain,
                                                 const float* __restrict__ w,
                                                 const float* __restrict__ bias,
                                                 float* __restrict__ out) {
  const int lane = threadIdx.x & 31;
  const int wid  = threadIdx.x >> 5;
  const int gw   = blockIdx.x * 4 + wid;
  const int MT = (B_ * N_) / 16;  // 512
  const int CT = C_ / 64;         // 12
  if (gw >= MT * CT) return;
  const int mt = gw % MT, ct = gw / MT;
  const int m0 = mt * 16, n0 = ct * 64;
  const int half = lane >> 4, l16 = lane & 15;

  v8f acc[4] = {};
  for (int k0 = 0; k0 < C_; k0 += 32) {
    const _Float16* ar = ain + (size_t)(m0 + l16) * C_ + k0 + half * 16;
    v16h a;
#pragma unroll
    for (int i = 0; i < 8; ++i) { a[i] = ar[i]; a[i + 8] = ar[i + 8]; }
#pragma unroll
    for (int t = 0; t < 4; ++t) {
      const float* br = w + (size_t)(n0 + t * 16 + l16) * C_ + k0 + half * 16;
      v16h bfr;
#pragma unroll
      for (int i = 0; i < 16; ++i) bfr[i] = (_Float16)br[i];
      acc[t] = wmma_f16(a, bfr, acc[t]);
    }
  }
#pragma unroll
  for (int t = 0; t < 4; ++t) {
    const int col = n0 + t * 16 + l16;
    const float bb = bias[col];
#pragma unroll
    for (int r = 0; r < 8; ++r)
      out[(size_t)(m0 + r + 8 * half) * C_ + col] = acc[t][r] + bb;
  }
}

extern "C" void kernel_launch(void* const* d_in, const int* in_sizes, int n_in,
                              void* d_out, int out_size, void* d_ws, size_t ws_size,
                              hipStream_t stream) {
  (void)in_sizes; (void)n_in; (void)out_size; (void)ws_size;
  const float* x      = (const float*)d_in[0];
  const int*   mask   = (const int*)d_in[1];
  const float* qkv_w  = (const float*)d_in[2];
  const float* proj_w = (const float*)d_in[3];
  const float* proj_b = (const float*)d_in[4];
  float* out = (float*)d_out;

  const size_t BHND = (size_t)B_ * H_ * N_ * D_;   // 6,291,456
  _Float16* qkvh = (_Float16*)d_ws;                // q|k|v f16, 3*BHND halfs
  _Float16* aoh  = qkvh + 3 * BHND;                // attn out f16, B*N*C halfs

  {
    const int waves = ((B_ * N_) / 16) * ((3 * C_) / 64);  // 18432
    qkv_gemm<<<waves / 4, 128, 0, stream>>>(x, qkv_w, qkvh);
  }
  {
    const int waves = B_ * H_ * (N_ / 16);  // 6144
    attn_kernel<<<waves / 4, 128, 0, stream>>>(qkvh, qkvh + BHND,
                                               qkvh + 2 * BHND, mask, aoh);
  }
  {
    const int waves = ((B_ * N_) / 16) * (C_ / 64);  // 6144
    proj_gemm<<<waves / 4, 128, 0, stream>>>(aoh, proj_w, proj_b, out);
  }
}
